// Network_55301998903390
// MI455X (gfx1250) — matrix operand order repaired and checked by reading, not compiled, not run on hardware
//
#include <hip/hip_runtime.h>
#include <hip/hip_bf16.h>

#define CCH 128              // channels
#define KN  27               // kernel offsets
#define NPAR 4096
#define MCH  32768

typedef __attribute__((ext_vector_type(16))) _Float16 v16h;
typedef __attribute__((ext_vector_type(8)))  float    v8f;
typedef __attribute__((ext_vector_type(4)))  unsigned int u32x4;

union HFrag { v16h v; u32x4 q[2]; };

#define WMMA_F16(A, B, Cacc) \
  __builtin_amdgcn_wmma_f32_16x16x32_f16(false, (A), false, (B), (short)0, (Cacc), false, false)

// ---------------------------------------------------------------------------
// Pack fp32 weights [nk][128][128] into per-lane WMMA B-fragment layout:
//   out[((k*4+cc)*8+dt)*32 + lane][j]  (16 halves per lane, contiguous 32B)
// lane 0-15  -> column n=lane,    kk = 0..7, 16..23
// lane 16-31 -> column n=lane-16, kk = 8..15, 24..31
// ---------------------------------------------------------------------------
__global__ void pack_w_kernel(const float* __restrict__ W,
                              _Float16* __restrict__ out, int nk) {
  int idx = blockIdx.x * blockDim.x + threadIdx.x;
  int total = nk * 4 * 8 * 32 * 16;
  if (idx >= total) return;
  int j    = idx & 15;
  int lane = (idx >> 4) & 31;
  int dt   = (idx >> 9) & 7;
  int cc   = (idx >> 12) & 3;
  int k    = idx >> 14;
  int kk   = ((j >> 3) << 4) + ((lane >> 4) << 3) + (j & 7);
  int n    = lane & 15;
  int c    = cc * 32 + kk;
  int d    = dt * 16 + n;
  out[idx] = (_Float16)W[(size_t)k * CCH * CCH + (size_t)c * CCH + d];
}

// f0 = prior_emb[x_O], cast to f16
__global__ void embed_parent_kernel(const int* __restrict__ xO,
                                    const float* __restrict__ emb,
                                    _Float16* __restrict__ out, int P) {
  int idx = blockIdx.x * blockDim.x + threadIdx.x;
  if (idx >= P * CCH) return;
  int p = idx / CCH, c = idx % CCH;
  out[idx] = (_Float16)emb[(size_t)xO[p] * CCH + c];
}

// g0 = repeat(f3, 8) + oct_emb[child_oct]
__global__ void fcg_kernel(const _Float16* __restrict__ f,
                           const int* __restrict__ oct,
                           const float* __restrict__ oct_emb,
                           _Float16* __restrict__ g, int M) {
  int idx = blockIdx.x * blockDim.x + threadIdx.x;
  if (idx >= M * CCH) return;
  int m = idx / CCH, c = idx % CCH;
  g[idx] = (_Float16)((float)f[(size_t)(m >> 3) * CCH + c] +
                      oct_emb[(size_t)oct[m] * CCH + c]);
}

// g1 = g3 + s1_emb[gt % 16]
__global__ void addemb_kernel(const _Float16* __restrict__ g3,
                              const int* __restrict__ gt,
                              const float* __restrict__ s1_emb,
                              _Float16* __restrict__ g1, int M) {
  int idx = blockIdx.x * blockDim.x + threadIdx.x;
  if (idx >= M * CCH) return;
  int m = idx / CCH, c = idx % CCH;
  int s = gt[m] & 15;
  g1[idx] = (_Float16)((float)g3[idx] + s1_emb[(size_t)s * CCH + c]);
}

// ---------------------------------------------------------------------------
// Sparse submanifold conv via WMMA f16->f32, 64-point M-tile per block.
//   out[p, d] = relu( sum_k sum_c fin[nbr[p,k], c] * W[k,c,d]  (+ res[p,d]) )
// Block = 256 threads (8 wave32). Wave w owns out-col tile [16w, 16w+16) for
// all 4 point-subtiles (4 v8f accumulators). Per k: software-pipelined,
// double-buffered LDS gather (issue k+1 gather before computing k); per cc the
// four A fragments are loaded into distinct registers up-front so the 8
// ds_load_b128 cluster and overlap with the 4 WMMAs they feed.
// ---------------------------------------------------------------------------
__global__ void __launch_bounds__(256)
spconv_kernel(const _Float16* __restrict__ fin,   // [P,128] f16
              const int*      __restrict__ nbr,   // [P,27]
              const _Float16* __restrict__ Wp,    // packed fragments
              const _Float16* __restrict__ res,   // [P,128] f16 or null
              _Float16*       __restrict__ fout,  // [P,128] f16
              int P) {
  __shared__ _Float16 As[2][64][CCH];             // 32 KB double buffer
  const int pbase = blockIdx.x * 64;
  const int tid   = threadIdx.x;
  const int wave  = tid >> 5;                     // 0..7 -> out-col tile
  const int lane  = tid & 31;

  v8f acc0 = {}, acc1 = {}, acc2 = {}, acc3 = {};

  // gather mapping: thread loads one 16B segment from 4 rows (r, r+16, ...)
  const int grow = tid >> 4;                      // 0..15
  const int gseg = tid & 15;                      // 16B segment in a 256B row

  u32x4 gr[4];
#pragma unroll
  for (int t = 0; t < 4; ++t) {
    int r   = grow + 16 * t;
    int src = nbr[(size_t)(pbase + r) * KN + 0];
    gr[t]   = *(const u32x4*)(fin + (size_t)src * CCH + gseg * 8);
  }

  const int m  = lane & 15;
  const int hi = (lane >> 4) * 8;

  for (int k = 0; k < KN; ++k) {
    const int buf = k & 1;
#pragma unroll
    for (int t = 0; t < 4; ++t)
      *(u32x4*)&As[buf][grow + 16 * t][gseg * 8] = gr[t];

    if (k + 1 < KN) {                             // pipeline next gather
#pragma unroll
      for (int t = 0; t < 4; ++t) {
        int r   = grow + 16 * t;
        int src = nbr[(size_t)(pbase + r) * KN + (k + 1)];
        gr[t]   = *(const u32x4*)(fin + (size_t)src * CCH + gseg * 8);
      }
      // warm L1 for next k's B-fragment stream
      __builtin_prefetch(Wp + ((((size_t)(k + 1) * 4) * 8 + wave) * 32 + lane) * 16, 0, 3);
    }
    __syncthreads();                              // tile k visible; closes reuse hazard

#pragma unroll
    for (int cc = 0; cc < 4; ++cc) {
      HFrag b;
      const u32x4* bq = (const u32x4*)(Wp +
          ((((size_t)k * 4 + cc) * 8 + wave) * 32 + lane) * 16);
      b.q[0] = bq[0];
      b.q[1] = bq[1];

      HFrag a0, a1, a2, a3;                       // distinct regs: cluster ds_loads
      a0.q[0] = *(const u32x4*)&As[buf][m][cc * 32 + hi];
      a0.q[1] = *(const u32x4*)&As[buf][m][cc * 32 + 16 + hi];
      a1.q[0] = *(const u32x4*)&As[buf][16 + m][cc * 32 + hi];
      a1.q[1] = *(const u32x4*)&As[buf][16 + m][cc * 32 + 16 + hi];
      a2.q[0] = *(const u32x4*)&As[buf][32 + m][cc * 32 + hi];
      a2.q[1] = *(const u32x4*)&As[buf][32 + m][cc * 32 + 16 + hi];
      a3.q[0] = *(const u32x4*)&As[buf][48 + m][cc * 32 + hi];
      a3.q[1] = *(const u32x4*)&As[buf][48 + m][cc * 32 + 16 + hi];
      acc0 = WMMA_F16(a0.v, b.v, acc0);
      acc1 = WMMA_F16(a1.v, b.v, acc1);
      acc2 = WMMA_F16(a2.v, b.v, acc2);
      acc3 = WMMA_F16(a3.v, b.v, acc3);
    }
  }

  // D layout: VGPR r -> row (r + 8*(lane>=16)), col = lane%16
  const int n     = lane & 15;
  const int mbase = (lane >> 4) * 8;
  v8f accs[4] = {acc0, acc1, acc2, acc3};
#pragma unroll
  for (int t = 0; t < 4; ++t) {
#pragma unroll
    for (int r = 0; r < 8; ++r) {
      int mm = 16 * t + mbase + r;
      size_t o = (size_t)(pbase + mm) * CCH + wave * 16 + n;
      float v = accs[t][r];
      if (res) v += (float)res[o];
      v = v > 0.f ? v : 0.f;
      fout[o] = (_Float16)v;
    }
  }
}

// Dense y = relu(x @ W + b), W packed as 1-k fragment set; 64-row tile.
__global__ void __launch_bounds__(256)
gemm128_kernel(const _Float16* __restrict__ fin,
               const _Float16* __restrict__ Wp,
               const float*    __restrict__ bias,
               _Float16*       __restrict__ fout, int P) {
  __shared__ _Float16 As[64][CCH];                // 16 KB
  const int pbase = blockIdx.x * 64;
  const int tid   = threadIdx.x;
  const int wave  = tid >> 5;
  const int lane  = tid & 31;
  {
    int row = tid >> 4, seg = tid & 15;
#pragma unroll
    for (int t = 0; t < 4; ++t)
      *(u32x4*)&As[row + 16 * t][seg * 8] =
          *(const u32x4*)(fin + (size_t)(pbase + row + 16 * t) * CCH + seg * 8);
  }
  __syncthreads();

  v8f acc0 = {}, acc1 = {}, acc2 = {}, acc3 = {};
  const int m  = lane & 15;
  const int hi = (lane >> 4) * 8;
#pragma unroll
  for (int cc = 0; cc < 4; ++cc) {
    HFrag b;
    const u32x4* bq = (const u32x4*)(Wp +
        (((size_t)cc * 8 + wave) * 32 + lane) * 16);
    b.q[0] = bq[0];
    b.q[1] = bq[1];
    HFrag a0, a1, a2, a3;
    a0.q[0] = *(const u32x4*)&As[m][cc * 32 + hi];
    a0.q[1] = *(const u32x4*)&As[m][cc * 32 + 16 + hi];
    a1.q[0] = *(const u32x4*)&As[16 + m][cc * 32 + hi];
    a1.q[1] = *(const u32x4*)&As[16 + m][cc * 32 + 16 + hi];
    a2.q[0] = *(const u32x4*)&As[32 + m][cc * 32 + hi];
    a2.q[1] = *(const u32x4*)&As[32 + m][cc * 32 + 16 + hi];
    a3.q[0] = *(const u32x4*)&As[48 + m][cc * 32 + hi];
    a3.q[1] = *(const u32x4*)&As[48 + m][cc * 32 + 16 + hi];
    acc0 = WMMA_F16(a0.v, b.v, acc0);
    acc1 = WMMA_F16(a1.v, b.v, acc1);
    acc2 = WMMA_F16(a2.v, b.v, acc2);
    acc3 = WMMA_F16(a3.v, b.v, acc3);
  }

  const int n     = lane & 15;
  const int mbase = (lane >> 4) * 8;
  float bb = bias[wave * 16 + n];
  v8f accs[4] = {acc0, acc1, acc2, acc3};
#pragma unroll
  for (int t = 0; t < 4; ++t) {
#pragma unroll
    for (int r = 0; r < 8; ++r) {
      int mm = 16 * t + mbase + r;
      size_t o = (size_t)(pbase + mm) * CCH + wave * 16 + n;
      float v = accs[t][r] + bb;
      v = v > 0.f ? v : 0.f;
      fout[o] = (_Float16)v;
    }
  }
}

// logits = y1 @ W2 + b2 (16 cols), softmax, gather gt bit, accumulate bits.
__global__ void head_kernel(const _Float16* __restrict__ y1,
                            const float*    __restrict__ W2,   // [128,16]
                            const float*    __restrict__ b2,   // [16]
                            const int*      __restrict__ gt,   // [M]
                            int stage, float* acc, int M) {
  int tid = blockIdx.x * blockDim.x + threadIdx.x;
  int row = tid >> 4;
  int j   = tid & 15;
  if (row >= M) return;
  float s = b2[j];
  const _Float16* yr = y1 + (size_t)row * CCH;
#pragma unroll 4
  for (int c = 0; c < CCH; ++c) s += (float)yr[c] * W2[c * 16 + j];
  float mx = s;
  for (int off = 8; off > 0; off >>= 1) mx = fmaxf(mx, __shfl_xor(mx, off, 16));
  float e = expf(s - mx);
  float sum = e;
  for (int off = 8; off > 0; off >>= 1) sum += __shfl_xor(sum, off, 16);
  int g = gt[row];
  int target = (stage == 0) ? (g & 15) : (g >> 4);
  if (j == target) {
    float p = e / sum + 1e-10f;
    float bits = -log2f(p);
    bits = fminf(fmaxf(bits, 0.f), 50.f);
    atomicAdd(acc, bits);
  }
}

__global__ void zero_acc_kernel(float* acc) { acc[0] = 0.f; }
__global__ void finalize_kernel(const float* acc, float* out) {
  out[0] = acc[0] * (1.0f / (float)NPAR);
}

// ---------------------------------------------------------------------------
extern "C" void kernel_launch(void* const* d_in, const int* in_sizes, int n_in,
                              void* d_out, int out_size, void* d_ws, size_t ws_size,
                              hipStream_t stream) {
  const int*   x_O       = (const int*)  d_in[0];
  const int*   gt_O      = (const int*)  d_in[1];
  const int*   nbr_par   = (const int*)  d_in[2];
  const int*   nbr_child = (const int*)  d_in[3];
  const int*   child_oct = (const int*)  d_in[4];
  const float* prior_emb = (const float*)d_in[5];
  const float* convW[10] = {
      (const float*)d_in[6],  (const float*)d_in[7],  (const float*)d_in[8],
      (const float*)d_in[9],  (const float*)d_in[10],                       // pW*
      (const float*)d_in[12], (const float*)d_in[13], (const float*)d_in[14],
      (const float*)d_in[15], (const float*)d_in[16]};                      // tW*
  const float* oct_emb = (const float*)d_in[11];
  const float* s1_emb  = (const float*)d_in[17];
  const float* h0W1 = (const float*)d_in[18];
  const float* h0b1 = (const float*)d_in[19];
  const float* h0W2 = (const float*)d_in[20];
  const float* h0b2 = (const float*)d_in[21];
  const float* h1W1 = (const float*)d_in[22];
  const float* h1b1 = (const float*)d_in[23];
  const float* h1W2 = (const float*)d_in[24];
  const float* h1b2 = (const float*)d_in[25];

  // workspace carve-up (256B aligned)
  char* ws = (char*)d_ws;
  auto carve = [&](size_t bytes) -> void* {
    void* p = (void*)ws;
    ws += (bytes + 255) & ~(size_t)255;
    return p;
  };
  const size_t PKW_ELEMS = (size_t)KN * 4 * 8 * 32 * 16;   // per conv
  const size_t DKW_ELEMS = (size_t)4 * 8 * 32 * 16;        // per dense
  _Float16* wp[10];
  for (int i = 0; i < 10; ++i) wp[i] = (_Float16*)carve(PKW_ELEMS * 2);
  _Float16* dwp[2];
  for (int i = 0; i < 2; ++i) dwp[i] = (_Float16*)carve(DKW_ELEMS * 2);
  _Float16* pb0 = (_Float16*)carve((size_t)NPAR * CCH * 2);
  _Float16* pb1 = (_Float16*)carve((size_t)NPAR * CCH * 2);
  _Float16* pb2 = (_Float16*)carve((size_t)NPAR * CCH * 2);
  _Float16* cb0 = (_Float16*)carve((size_t)MCH * CCH * 2);
  _Float16* cb1 = (_Float16*)carve((size_t)MCH * CCH * 2);
  _Float16* cb2 = (_Float16*)carve((size_t)MCH * CCH * 2);
  _Float16* cb3 = (_Float16*)carve((size_t)MCH * CCH * 2);
  float*    acc = (float*)carve(256);

  // ---- weight packing ----
  {
    int total = (int)PKW_ELEMS;
    int blk = (total + 255) / 256;
    for (int i = 0; i < 10; ++i)
      pack_w_kernel<<<blk, 256, 0, stream>>>(convW[i], wp[i], KN);
    int dtotal = (int)DKW_ELEMS;
    int dblk = (dtotal + 255) / 256;
    pack_w_kernel<<<dblk, 256, 0, stream>>>(h0W1, dwp[0], 1);
    pack_w_kernel<<<dblk, 256, 0, stream>>>(h1W1, dwp[1], 1);
  }

  // ---- parent branch ----
  embed_parent_kernel<<<(NPAR * CCH + 255) / 256, 256, 0, stream>>>(
      x_O, prior_emb, pb0, NPAR);
  spconv_kernel<<<NPAR / 64, 256, 0, stream>>>(pb0, nbr_par, wp[0], nullptr, pb1, NPAR);
  spconv_kernel<<<NPAR / 64, 256, 0, stream>>>(pb1, nbr_par, wp[1], nullptr, pb2, NPAR);
  spconv_kernel<<<NPAR / 64, 256, 0, stream>>>(pb2, nbr_par, wp[2], pb1,    pb0, NPAR); // f2
  spconv_kernel<<<NPAR / 64, 256, 0, stream>>>(pb0, nbr_par, wp[3], nullptr, pb2, NPAR);
  spconv_kernel<<<NPAR / 64, 256, 0, stream>>>(pb2, nbr_par, wp[4], pb0,    pb1, NPAR); // f3

  // ---- children branch ----
  fcg_kernel<<<(MCH * CCH + 255) / 256, 256, 0, stream>>>(
      pb1, child_oct, oct_emb, cb0, MCH);                                   // g0
  spconv_kernel<<<MCH / 64, 256, 0, stream>>>(cb0, nbr_child, wp[5], nullptr, cb1, MCH);
  spconv_kernel<<<MCH / 64, 256, 0, stream>>>(cb1, nbr_child, wp[6], nullptr, cb2, MCH);
  spconv_kernel<<<MCH / 64, 256, 0, stream>>>(cb2, nbr_child, wp[7], cb1,    cb3, MCH); // g2
  spconv_kernel<<<MCH / 64, 256, 0, stream>>>(cb3, nbr_child, wp[8], nullptr, cb2, MCH);
  spconv_kernel<<<MCH / 64, 256, 0, stream>>>(cb2, nbr_child, wp[9], cb3,    cb1, MCH); // g3

  // ---- heads ----
  zero_acc_kernel<<<1, 1, 0, stream>>>(acc);
  gemm128_kernel<<<MCH / 64, 256, 0, stream>>>(cb1, dwp[0], h0b1, cb0, MCH); // y1_0
  head_kernel<<<(MCH * 16 + 255) / 256, 256, 0, stream>>>(
      cb0, h0W2, h0b2, gt_O, 0, acc, MCH);
  addemb_kernel<<<(MCH * CCH + 255) / 256, 256, 0, stream>>>(
      cb1, gt_O, s1_emb, cb2, MCH);                                          // g1
  gemm128_kernel<<<MCH / 64, 256, 0, stream>>>(cb2, dwp[1], h1b1, cb3, MCH); // y1_1
  head_kernel<<<(MCH * 16 + 255) / 256, 256, 0, stream>>>(
      cb3, h1W2, h1b2, gt_O, 1, acc, MCH);
  finalize_kernel<<<1, 1, 0, stream>>>(acc, (float*)d_out);
}
